// LuongAttention_9165460209993
// MI455X (gfx1250) — matrix operand order repaired and checked by reading, not compiled, not run on hardware
//
#include <hip/hip_runtime.h>
#include <hip/hip_bf16.h>
#include <math.h>

// ---------------------------------------------------------------------------
// Problem: S=2048, B=16, H=1024
//   base[b,h]  = b_att[h] + hidden[b,:] . W_att[h, 0:H]          (tiny GEMM)
//   pre[s,b,h] = base[b,h] + hseq[s,b,:] . W_att[h, H:2H]        (big GEMM)
//   score[s,b] = sum_h v[h]*tanh(pre[s,b,h])                     (fused)
//   attn       = softmax over b (faithful to reference axis=1)
//   ctx[b,h]   = sum_s attn[s,b]*hseq[s,b,h]
// Roofline: 69 GFLOP bf16-WMMA GEMM + 2x134MB hseq streaming (~12us HBM floor)
// ---------------------------------------------------------------------------

#define S_LEN 2048
#define BATCH 16
#define HID   1024

#define ROWS_PER_WG 64                      // 4 s-positions * 16 batches
#define S_PER_WG    (ROWS_PER_WG / BATCH)   // 4 M-tiles per WG

// LDS layout for the A slab: one block per m16 = row%16 (the lane's row id),
// holding that lane's 4 mt-rows contiguously. All inner-loop DS offsets
// (mt*MT_STRIDE + ks*64 [+32] <= 8208B) become 16-bit immediates off ONE base.
// Bank spread: LN_STRIDE = 2076 dwords, 2076 % 64 = 28, and {28*ln mod 64}
// covers all multiples of 4 -> lanes 0-15 tile the 64 banks conflict-free.
#define MT_STRIDE   2064                    // bytes per row (1024 bf16 + 16B pad)
#define LN_STRIDE   (4 * MT_STRIDE + 48)    // 8304 bytes per m16 block
#define A_LDS_BYTES (16 * LN_STRIDE)        // 132864
#define SC_LDS_OFF  A_LDS_BYTES
#define LDS_TOTAL   (A_LDS_BYTES + ROWS_PER_WG * 4)

typedef __attribute__((ext_vector_type(16))) __bf16 v16bf;
typedef __attribute__((ext_vector_type(8)))  float  v8f;
typedef __attribute__((ext_vector_type(4)))  float  v4f;

union FragU { struct { uint4 lo; uint4 hi; } q; v16bf v; };

__device__ __forceinline__ v16bf make_frag(uint4 lo, uint4 hi) {
  FragU u; u.q.lo = lo; u.q.hi = hi; return u.v;
}

// ---- native V_TANH_F32 when the builtin exists; branch-free fallback ------
#if defined(__has_builtin)
#if __has_builtin(__builtin_amdgcn_tanhf)
#define FAST_TANH(x) __builtin_amdgcn_tanhf(x)
#elif __has_builtin(__builtin_amdgcn_tanh_f32)
#define FAST_TANH(x) __builtin_amdgcn_tanh_f32(x)
#endif
#endif
#ifndef FAST_TANH
__device__ __forceinline__ float fast_tanh_impl(float x) {
  // tanh(x) = 1 - 2/(exp2(2*log2e*x) + 1); exact limits at +-inf, no branches
  float e = __builtin_amdgcn_exp2f(x * 2.8853900817779268f);
  return 1.0f - 2.0f / (e + 1.0f);
}
#define FAST_TANH(x) fast_tanh_impl(x)
#endif

// ---------------- K0: W2 (strided fp32) -> dense bf16, L2-resident ---------
__global__ void w2_cvt_kernel(const float* __restrict__ W, __bf16* __restrict__ w2) {
  int id = blockIdx.x * blockDim.x + threadIdx.x;   // 1M elements
  int h = id >> 10, k = id & (HID - 1);
  w2[id] = (__bf16)W[(size_t)h * (2 * HID) + HID + k];
}

// ---------------- K1: baseT[h,b] (fp32, transposed for epilogue) -----------
__global__ void base_kernel(const float* __restrict__ hidden,
                            const float* __restrict__ W,
                            const float* __restrict__ b_att,
                            float* __restrict__ baseT) {
  int id = blockIdx.x * blockDim.x + threadIdx.x;   // 16384 = b*1024 + h
  int b = id >> 10, h = id & (HID - 1);
  const float* hp = hidden + (size_t)b * HID;
  const float* wp = W + (size_t)h * (2 * HID);
  float acc = b_att[h];
  for (int k = 0; k < HID; k += 4) {
    v4f hv = *(const v4f*)(hp + k);
    v4f wv = *(const v4f*)(wp + k);
    acc += hv.x * wv.x + hv.y * wv.y + hv.z * wv.z + hv.w * wv.w;
  }
  baseT[h * BATCH + b] = acc;   // transposed: 8 consecutive floats per lane
}

// ---------------- K2: fused GEMM + tanh + v-dot -> score -------------------
extern __shared__ char lds_raw[];

__global__ __launch_bounds__(256, 2)   // cap VGPRs so 2 waves/SIMD fit
void score_gemm_kernel(const float* __restrict__ hseq,
                       const __bf16* __restrict__ w2,
                       const float* __restrict__ baseT,
                       const float* __restrict__ v,
                       float* __restrict__ score) {
  char*  aTile    = lds_raw;
  float* ldsScore = (float*)(lds_raw + SC_LDS_OFF);

  const int tid  = threadIdx.x;
  const int wg   = blockIdx.x;        // 512 WGs, each owns 64 consecutive rows
  const int wave = tid >> 5;
  const int lane = tid & 31;
  const int ln   = lane & 15;
  const int hi   = lane >> 4;

  // ---- stage A slab: 64x1024 fp32 (contiguous) -> bf16 LDS (permuted) -----
  // non-temporal: hseq is streamed once here, keep W2/base resident in L2
  const float* g = hseq + (size_t)wg * ROWS_PER_WG * HID;
  for (int i = tid * 4; i < ROWS_PER_WG * HID; i += 256 * 4) {
    int row = i >> 10;
    int col = i & (HID - 1);
    v4f f = __builtin_nontemporal_load((const v4f*)(g + i));
    __bf16* d = (__bf16*)(aTile + (row & 15) * LN_STRIDE +
                          (row >> 4) * MT_STRIDE + col * 2);
    d[0] = (__bf16)f.x; d[1] = (__bf16)f.y;
    d[2] = (__bf16)f.z; d[3] = (__bf16)f.w;
  }
  __syncthreads();

  float sc[S_PER_WG][8];
#pragma unroll
  for (int mt = 0; mt < S_PER_WG; ++mt)
#pragma unroll
    for (int r = 0; r < 8; ++r) sc[mt][r] = 0.0f;

  // Single A base pointer per lane; every fragment chunk is base + immediate.
  // A frag per ISA layout: lane holds row m = mt*16+ln,
  // K chunks {+0..7,+16..23} (lanes 0-15) / {+8..15,+24..31} (lanes 16-31).
  const char* ap_base = aTile + ln * LN_STRIDE + hi * 16;

  // each wave owns 8 contiguous N-tiles (16 h-columns each)
  for (int j = 0; j < 8; ++j) {
    const int nt = wave * 8 + j;
    const int h0 = nt * 16;
    const float vv = v[h0 + ln];

    // baseT[(h0+ln)*16 + 8*hi + r], r=0..7 : two coalesced b128 loads
    const float* bptr = baseT + (h0 + ln) * BATCH + 8 * hi;
    v4f blo = *(const v4f*)(bptr);
    v4f bhi = *(const v4f*)(bptr + 4);
    float bs[8] = {blo.x, blo.y, blo.z, blo.w, bhi.x, bhi.y, bhi.z, bhi.w};

    v8f acc[S_PER_WG];
#pragma unroll
    for (int mt = 0; mt < S_PER_WG; ++mt)
      acc[mt] = (v8f){0.f,0.f,0.f,0.f,0.f,0.f,0.f,0.f};

    // B frag per ISA layout: lane ln holds column n=ln, K chunks
    // ks*32 + 8*hi + {0..7, 16..23}  (row-major bf16 W2, stride 1024)
    const __bf16* bp = w2 + (size_t)(h0 + ln) * HID + hi * 8;

#pragma unroll 2
    for (int ks = 0; ks < 32; ++ks) {
      uint4 b0 = *(const uint4*)(bp + ks * 32);
      uint4 b1 = *(const uint4*)(bp + ks * 32 + 16);
      v16bf bfrag = make_frag(b0, b1);

#pragma unroll
      for (int mt = 0; mt < S_PER_WG; ++mt) {
        const char* ap = ap_base + mt * MT_STRIDE + ks * 64;
        v16bf afrag = make_frag(*(const uint4*)ap, *(const uint4*)(ap + 32));
        acc[mt] = __builtin_amdgcn_wmma_f32_16x16x32_bf16(
            false, afrag, false, bfrag, (short)0, acc[mt], false, false);
      }
    }

    // epilogue: C element (m=r+8*hi, n=ln) -> v[h]*tanh(pre); sum over n later
#pragma unroll
    for (int mt = 0; mt < S_PER_WG; ++mt)
#pragma unroll
      for (int r = 0; r < 8; ++r)
        sc[mt][r] += vv * FAST_TANH(acc[mt][r] + bs[r]);
  }

  // reduce over the 16 columns held in each 16-lane half
#pragma unroll
  for (int mt = 0; mt < S_PER_WG; ++mt)
#pragma unroll
    for (int r = 0; r < 8; ++r) {
      float x = sc[mt][r];
      x += __shfl_xor(x, 1);
      x += __shfl_xor(x, 2);
      x += __shfl_xor(x, 4);
      x += __shfl_xor(x, 8);
      sc[mt][r] = x;
    }

  __syncthreads();
  if (tid < ROWS_PER_WG) ldsScore[tid] = 0.0f;
  __syncthreads();
  if (ln == 0) {
#pragma unroll
    for (int mt = 0; mt < S_PER_WG; ++mt)
#pragma unroll
      for (int r = 0; r < 8; ++r)
        atomicAdd(&ldsScore[mt * 16 + r + 8 * hi], sc[mt][r]);  // ds_add_f32
  }
  __syncthreads();
  if (tid < ROWS_PER_WG)
    score[(size_t)wg * ROWS_PER_WG + tid] = ldsScore[tid];
}

// ---------------- K3: masked softmax over batch axis (axis=1) --------------
__global__ void softmax_kernel(const float* __restrict__ score,
                               const int* __restrict__ masks,
                               float* __restrict__ attn) {
  int s = blockIdx.x * blockDim.x + threadIdx.x;
  if (s >= S_LEN) return;
  float vals[BATCH];
  float mx = -INFINITY;
#pragma unroll
  for (int j = 0; j < BATCH; ++j) {
    float x = masks[s * BATCH + j] ? score[s * BATCH + j] : -1.0e10f;
    vals[j] = x;
    mx = fmaxf(mx, x);
  }
  float sum = 0.0f;
#pragma unroll
  for (int j = 0; j < BATCH; ++j) {
    float e = __expf(vals[j] - mx);
    vals[j] = e;
    sum += e;
  }
  float inv = 1.0f / sum;
#pragma unroll
  for (int j = 0; j < BATCH; ++j) attn[s * BATCH + j] = vals[j] * inv;
}

// ---------------- K4: context[b,h] = sum_s attn[s,b]*hseq[s,b,h] -----------
__global__ __launch_bounds__(64)
void context_kernel(const float* __restrict__ hseq,
                    const float* __restrict__ attn,
                    float* __restrict__ out) {
  int b = blockIdx.y;
  int h = blockIdx.x * 64 + threadIdx.x;
  const float* p = hseq + (size_t)b * HID + h;
  float acc = 0.0f;
#pragma unroll 4
  for (int s = 0; s < S_LEN; ++s)
    acc += attn[s * BATCH + b] *
           __builtin_nontemporal_load(p + (size_t)s * BATCH * HID);
  out[b * HID + h] = acc;
}

// ---------------------------------------------------------------------------
extern "C" void kernel_launch(void* const* d_in, const int* in_sizes, int n_in,
                              void* d_out, int out_size, void* d_ws, size_t ws_size,
                              hipStream_t stream) {
  const float* hidden = (const float*)d_in[0];
  const float* hseq   = (const float*)d_in[1];
  const int*   masks  = (const int*)d_in[2];
  const float* W      = (const float*)d_in[3];
  const float* b_att  = (const float*)d_in[4];
  const float* v      = (const float*)d_in[5];
  float* out = (float*)d_out;

  // workspace layout (total ~2.4 MB)
  char* ws = (char*)d_ws;
  __bf16* w2    = (__bf16*)ws;                                   // 2 MB
  float*  baseT = (float*)(ws + (2u << 20));                     // 64 KB
  float*  scor  = (float*)(ws + (2u << 20) + (64u << 10));       // 128 KB
  float*  attn  = (float*)(ws + (2u << 20) + (192u << 10));      // 128 KB

  w2_cvt_kernel<<<(HID * HID) / 256, 256, 0, stream>>>(W, w2);
  base_kernel<<<(BATCH * HID) / 256, 256, 0, stream>>>(hidden, W, b_att, baseT);
  score_gemm_kernel<<<(S_LEN * BATCH) / ROWS_PER_WG, 256, LDS_TOTAL, stream>>>(
      hseq, w2, baseT, v, scor);
  softmax_kernel<<<S_LEN / 256, 256, 0, stream>>>(scor, masks, attn);
  context_kernel<<<dim3(HID / 64, BATCH), 64, 0, stream>>>(hseq, attn, out);
}
